// DifferentiableSkeletonize_58703613002527
// MI455X (gfx1250) — matrix operand order repaired and checked by reading, not compiled, not run on hardware
//
#include <hip/hip_runtime.h>
#include <hip/hip_bf16.h>
#include <cstdint>

// DifferentiableSkeletonize, fused single pass for gfx1250 (MI455X).
// dilated = avg3(avg3(x)) = separable 5-tap [1,2,3,2,1]/9 per axis, zero pad.
// Bandwidth-bound: >=268 MB mandatory @ 23.3 TB/s ~= 11.5 us floor. One fused
// pass, 2.5-D rolling planes in LDS staged by the CDNA5 async-LDS DMA path
// (software-pipelined one plane deep, counted s_wait_asynccnt); d-axis conv =
// banded GEMM on the f16 WMMA unit with integer taps (exact: mask is 0/1,
// taps {1,2,3,2,1} exact in f16, f32 accumulate; 1/729 applied once in f32).
// Two workgroup barriers per output plane.

#ifndef __has_builtin
#define __has_builtin(x) 0
#endif
#if __has_builtin(__builtin_amdgcn_global_load_async_to_lds_b32)
#define USE_ASYNC_LDS 1
#endif

#ifdef USE_ASYNC_LDS
#if __has_builtin(__builtin_amdgcn_s_wait_asynccnt)
#define WAIT_ASYNC(n) __builtin_amdgcn_s_wait_asynccnt(n)
#else
#define WAIT_ASYNC(n) asm volatile("s_wait_asynccnt " #n ::: "memory")
#endif
#else
#define WAIT_ASYNC(n) do {} while (0)
#endif

typedef __attribute__((ext_vector_type(16))) _Float16 v16h;
typedef __attribute__((ext_vector_type(8)))  float    v8f;

#define Hh 256
#define Ww 256
#define Dd 256
#define TILE 32
#define HALO 2
#define RAW_W 36       // TILE + 2*HALO rows per plane
#define RAW_PITCH 52   // mult. of 4 (16B rows); 52*r mod 64 distinct r=0..15
#define PLANE_ELEMS (RAW_W * RAW_W)          // 1296 real elements per staged plane
#define T_ROWS 48      // 3 WMMA row-tiles (rows 36..47 dead)
#define T_PITCH 34     // 8*34 == 16 (mod 64): C-store halves hit disjoint banks
#define CHUNK 64

struct alignas(16) SMem {
  float raw[5][RAW_W][RAW_PITCH];  // staged planes (circular by plane%5); cols 36..51 = 0
  float tbuf[T_ROWS][T_PITCH];     // d-conv partial (integer-valued 0..9)
  float qbuf[5][TILE][TILE];       // (w,d)-conv planes (integer 0..81), circular
};

__device__ __forceinline__ int slotOf(int p) { int s = p % 5; return s < 0 ? s + 5 : s; }

#ifdef USE_ASYNC_LDS
__device__ __forceinline__ void async_cp_b32(const float* g, float* l) {
  __builtin_amdgcn_global_load_async_to_lds_b32((int*)(void*)g, (int*)(void*)l, 0, 0);
}
#endif

// per-thread staging address tuples (plane-invariant; computed once)
struct StageCtx {
  int  gofs[6];    // element offset inside a (w,d) plane (clamped, always valid)
  int  lofs[6];    // LDS float offset inside a raw slot: ww*RAW_PITCH+dd
  bool inb[6];     // (w,d) in bounds?
};

__device__ __forceinline__ StageCtx make_ctx(int tid, int w0, int d0) {
  StageCtx c;
#pragma unroll
  for (int i = 0; i < 6; ++i) {
    int ee = tid + 256 * i;
    if (ee >= PLANE_ELEMS) ee = tid + (PLANE_ELEMS - 256);  // tail: distinct addrs
    int ww = ee / RAW_W;
    int dd = ee - RAW_W * ww;
    int gw = w0 - HALO + ww;
    int gd = d0 - HALO + dd;
    c.inb[i]  = ((unsigned)gw < (unsigned)Ww) && ((unsigned)gd < (unsigned)Dd);
    c.gofs[i] = min(max(gw, 0), Ww - 1) * Dd + min(max(gd, 0), Dd - 1);
    c.lofs[i] = ww * RAW_PITCH + dd;
  }
  return c;
}

// ---- stage one 36x36 halo plane (h-index p): exactly 6 async ops per wave ----
__device__ __forceinline__ void stage_plane(const float* __restrict__ src,
                                            const float* __restrict__ zero,
                                            int n, int p, const StageCtx& c, SMem& sm) {
  const int sl  = slotOf(p);
  const bool hv = (p >= 0) && (p < Hh);
  const float* plane = src + (size_t)(n * Hh + (hv ? p : 0)) * (size_t)(Ww * Dd);
  float* lbase = (float*)sm.raw + sl * (RAW_W * RAW_PITCH);
#pragma unroll
  for (int i = 0; i < 6; ++i) {
    bool use = hv && c.inb[i];
#ifdef USE_ASYNC_LDS
    const float* g = use ? (plane + c.gofs[i]) : zero;   // OOB lanes pull a 0.0f word
    async_cp_b32(g, lbase + c.lofs[i]);                  // unconditional: keeps count==6
#else
    float v = plane[c.gofs[i]];
    lbase[c.lofs[i]] = use ? v : 0.f;
#endif
  }
}

// ---- constant banded B (32x16 f16, K-major): B[k][jj] = max(0, 3-|k-jj-2|) ----
__device__ __forceinline__ v16h make_B(int tid) {
  const int lane = tid & 31;
  const int jj = lane & 15;
  const int hi = lane >> 4;
  v16h b;
#pragma unroll
  for (int i = 0; i < 16; ++i) {
    int k = i + 8 * hi + (i >= 8 ? 8 : 0);    // table order: i<8 K=8*hi+i; i>=8 K=16+8*hi+(i-8)
    float v = fmaxf(0.f, 3.f - fabsf((float)(k - jj - 2)));
    b[i] = (_Float16)v;
  }
  return b;
}

// ---- pass1: d-axis 5-tap conv via v_wmma_f32_16x16x32_f16; one tile per wave ----
__device__ __forceinline__ void pass1_wmma(int p, const v16h B, SMem& sm) {
  const int tid  = threadIdx.x;
  const int wave = tid >> 5;
  if (wave >= 6) return;                      // uniform per-wave branch (EXEC stays all-1)
  const int lane = tid & 31;
  const int m  = lane & 15;
  const int hi = lane >> 4;
  const int sl = slotOf(p);
  const int mbase = (wave >> 1) * 16;         // 0,16,32
  const int dbase = (wave & 1) * 16;          // 0,16
  const int row = mbase + m;                  // 0..47 (rows>=36 feed dead tbuf rows)

  const float* rp = (const float*)sm.raw + (size_t)(sl * RAW_W + row) * RAW_PITCH
                  + dbase + 8 * hi;
  float4 r0a = *(const float4*)(rp + 0);
  float4 r0b = *(const float4*)(rp + 4);
  float4 r1a = *(const float4*)(rp + 16);
  float4 r1b = *(const float4*)(rp + 20);

  v16h a;
  a[0]  = (_Float16)r0a.x; a[1]  = (_Float16)r0a.y; a[2]  = (_Float16)r0a.z; a[3]  = (_Float16)r0a.w;
  a[4]  = (_Float16)r0b.x; a[5]  = (_Float16)r0b.y; a[6]  = (_Float16)r0b.z; a[7]  = (_Float16)r0b.w;
  a[8]  = (_Float16)r1a.x; a[9]  = (_Float16)r1a.y; a[10] = (_Float16)r1a.z; a[11] = (_Float16)r1a.w;
  a[12] = (_Float16)r1b.x; a[13] = (_Float16)r1b.y; a[14] = (_Float16)r1b.z; a[15] = (_Float16)r1b.w;

  v8f c = {};
  c = __builtin_amdgcn_wmma_f32_16x16x32_f16(false, a, false, B, (short)0, c,
                                             false, false);
#pragma unroll
  for (int r = 0; r < 8; ++r)                 // C: VGPR r = rows r (lo) / r+8 (hi)
    sm.tbuf[mbase + r + 8 * hi][dbase + m] = c[r];
}

// ---- pass2 (prologue only): w-axis 5-tap, tbuf -> qbuf[slot] ----------------
__device__ __forceinline__ void pass2(int p, SMem& sm) {
  const int tid = threadIdx.x;
  const int d  = tid & 31;
  const int wb = tid >> 5;
  const int sl = slotOf(p);
#pragma unroll
  for (int e = 0; e < 4; ++e) {
    int w = wb + 8 * e;
    float q = sm.tbuf[w + 0][d] + 2.f * sm.tbuf[w + 1][d] + 3.f * sm.tbuf[w + 2][d]
            + 2.f * sm.tbuf[w + 3][d] +       sm.tbuf[w + 4][d];
    sm.qbuf[sl][w][d] = q;
  }
}

// ---- fused pass2+emit: compute q[h+2] (reg), store it, combine, tail, store ----
__device__ __forceinline__ void pass2_emit(float* __restrict__ out, int n, int h,
                                           int w0, int d0, SMem& sm) {
  const int tid = threadIdx.x;
  const int d  = tid & 31;
  const int wb = tid >> 5;
  const int s0 = slotOf(h - 2), s1 = slotOf(h - 1), s2 = slotOf(h),
            s3 = slotOf(h + 1), s4 = slotOf(h + 2);
  float* oplane = out + (size_t)(n * Hh + h) * (size_t)(Ww * Dd)
                      + (size_t)w0 * Dd + d0;
#pragma unroll
  for (int e = 0; e < 4; ++e) {
    int w = wb + 8 * e;
    // w-axis 5-tap for plane h+2 (same (w,d) this thread needs below)
    float q4 = sm.tbuf[w + 0][d] + 2.f * sm.tbuf[w + 1][d] + 3.f * sm.tbuf[w + 2][d]
             + 2.f * sm.tbuf[w + 3][d] +       sm.tbuf[w + 4][d];
    sm.qbuf[s4][w][d] = q4;                   // visible to iters h+1.. via their barriers

    float x = sm.raw[s2][w + HALO][d + HALO];
    float acc = sm.qbuf[s0][w][d] + 2.f * sm.qbuf[s1][w][d] + 3.f * sm.qbuf[s2][w][d]
              + 2.f * sm.qbuf[s3][w][d] + q4;
    float dil  = acc * (1.0f / 729.0f);
    float diff = x - dil;
    float z    = diff * (1.0f / 0.3f) - (0.5f / 0.3f);
    float sig  = 1.0f / (1.0f + __expf(-z));
    float skel = 5.0f * (sig + diff);
    float o    = 0.3f * x + 0.7f * skel;
    o = fminf(fmaxf(o, 0.f), 1.f);
    o = rintf(o);                             // round-half-even == jnp.round
    oplane[w * Dd + d] = 1.0f - o;            // lanes 0..31 -> 128B contiguous store
  }
}

__global__ __launch_bounds__(256)
void DifferentiableSkeletonize_58703613002527_kernel(const float* __restrict__ src,
                                                     float* __restrict__ out,
                                                     const float* __restrict__ zero) {
  __shared__ SMem sm;
  const int tid = threadIdx.x;
  const int bx = blockIdx.x;                  // 64 (w,d) tiles
  const int w0 = (bx >> 3) * TILE;
  const int d0 = (bx & 7) * TILE;
  const int h0 = blockIdx.y * CHUNK;
  const int n  = blockIdx.z;

  const v16h B = make_B(tid);
  const StageCtx ctx = make_ctx(tid, w0, d0);

  // zero the A-tail pad columns once (cols 36..51; read by WMMA with zero weight)
  for (int i = tid; i < 5 * RAW_W * 16; i += 256) {
    int s = i / (RAW_W * 16);
    int r = (i >> 4) % RAW_W;
    int c = 36 + (i & 15);
    sm.raw[s][r][c] = 0.f;
  }

  // prologue: issue planes h0-2..h0+2 (5 x 6 async ops/wave), need first 4 resident
  for (int p = h0 - 2; p <= h0 + 2; ++p) stage_plane(src, zero, n, p, ctx, sm);
  WAIT_ASYNC(6);                              // newest plane (h0+2) may still fly
  __syncthreads();
  for (int p = h0 - 2; p <= h0 + 1; ++p) {
    pass1_wmma(p, B, sm);
    __syncthreads();
    pass2(p, sm);
    __syncthreads();
  }

  // main loop, pipelined one plane deep, 2 barriers per plane:
  //   issue h+3 into dead slot (h-2)%5; wait<=6 -> h+2 resident; barrier;
  //   WMMA d-conv of h+2 -> tbuf; barrier; fused w-conv + h-combine + tail.
  //   (tbuf WAR between iter h-1's fused phase and iter h's pass1 is protected
  //    by the post-wait barrier.)
  for (int h = h0; h < h0 + CHUNK; ++h) {
    stage_plane(src, zero, n, h + 3, ctx, sm);
    WAIT_ASYNC(6);
    __syncthreads();
    pass1_wmma(h + 2, B, sm);
    __syncthreads();
    pass2_emit(out, n, h, w0, d0, sm);
  }
}

extern "C" void kernel_launch(void* const* d_in, const int* in_sizes, int n_in,
                              void* d_out, int out_size, void* d_ws, size_t ws_size,
                              hipStream_t stream) {
  const float* seg = (const float*)d_in[0];
  float* out = (float*)d_out;
  // zero landing pad for out-of-bounds async lanes (graph-capture safe)
  hipMemsetAsync(d_ws, 0, 256, stream);
  const int N = in_sizes[0] / (Hh * Ww * Dd);     // batch*channels (=2)
  dim3 grid(64, Hh / CHUNK, N);                   // 64 tiles x 4 h-chunks x 2 = 512 WGs
  DifferentiableSkeletonize_58703613002527_kernel<<<grid, 256, 0, stream>>>(
      seg, out, (const float*)d_ws);
}
